// MultimodalSNN_9182640079529
// MI455X (gfx1250) — matrix operand order repaired and checked by reading, not compiled, not run on hardware
//
#include <hip/hip_runtime.h>

// ---------- types ----------
typedef __attribute__((ext_vector_type(16))) __bf16            v16bf;
typedef __attribute__((ext_vector_type(8)))  float             v8f;
typedef __attribute__((ext_vector_type(8)))  unsigned short    us8;

union Frag { v16bf v; us8 h[2]; };

__device__ __forceinline__ __bf16 f2bf(float f) {
    unsigned u = __builtin_bit_cast(unsigned, f);
    unsigned r = (u + 0x7FFFu + ((u >> 16) & 1u)) >> 16;
    unsigned short s = (unsigned short)r;
    return __builtin_bit_cast(__bf16, s);
}
__device__ __forceinline__ float bf2f(__bf16 b) {
    unsigned short s = __builtin_bit_cast(unsigned short, b);
    unsigned u = ((unsigned)s) << 16;
    return __builtin_bit_cast(float, u);
}

// CDNA5 async global->LDS copy: 16B per lane, ASYNCcnt-tracked.
__device__ __forceinline__ void async_cp16(__bf16* ldsDst, const __bf16* gSrc) {
    unsigned ldsOff = (unsigned)(unsigned long long)ldsDst;   // low 32b of generic = LDS offset
    unsigned long long ga = (unsigned long long)gSrc;
    asm volatile("global_load_async_to_lds_b128 %0, %1, off"
                 :: "v"(ldsOff), "v"(ga)
                 : "memory");
}
__device__ __forceinline__ void wait_async3() {   // all but 3 newest done
    asm volatile("s_wait_asynccnt 0x3" ::: "memory");
}
__device__ __forceinline__ void wait_async0() {   // all done
    asm volatile("s_wait_asynccnt 0x0" ::: "memory");
}

// ---------- model dims ----------
#define BB 4
#define TT 20
#define S_TEXT 64
#define S_AUDIO 32
#define DD 256
#define DSTATE 64
#define VOCAB 32000
#define AUDIO_F 128
#define HEADS 8
#define LL (TT * (S_TEXT + S_AUDIO))     // 1920
#define L_TEXT (TT * S_TEXT)             // 1280
#define HD (DD / HEADS)                  // 32

// =====================================================================
// bf16 WMMA GEMM: C = alpha*(A @ B^T_layout) + bias.   NO bounds checks:
// requires M%16==0, K%32==0, N%BN==0 (true for every launch here).
// A: [M,K] row-major (lda).  B: [N,K] row-major (ldb)  <- weights pre-transposed.
// Double-buffered async global->LDS pipeline; every wave issues exactly 3
// async instructions per stage so s_wait_asynccnt 3 is exact per wave.
// VSTORE=1: store bf16 output head-transposed Vt[((b*H+h)*HD+j)*LL + l].
// =====================================================================
template<int BN, int VSTORE>
__global__ void gemm_bf16_wmma(const __bf16* __restrict__ Aptr,
                               const __bf16* __restrict__ Bptr,
                               float* Cf, __bf16* Cbf, const float* bias,
                               int K,
                               int lda, int ldb, int ldc,
                               long long sA, long long sB, long long sC,
                               float alpha)
{
    constexpr int WAVES = BN / 16;
    constexpr int NT = 32 * WAVES;
    __shared__ __align__(16) __bf16 As[2][16][48];        // 16x32 tiles (96B rows)
    __shared__ __align__(16) __bf16 Bsh[2][BN * 48];      // BN x 32 tiles (96B rows)

    Aptr += (long long)blockIdx.z * sA;
    Bptr += (long long)blockIdx.z * sB;

    const int tid  = threadIdx.x;
    const int lane = tid & 31;
    const int wid  = tid >> 5;

    const int m0    = blockIdx.x * 16;
    const int n0blk = blockIdx.y * BN;

    const int m  = lane & 15;
    const int hs = lane >> 4;
    const int nl = wid * 16 + m;

    // staging assignments (uniform across all waves: 1 A chunk + 2 B chunks)
    const int ca = tid & 63;                 // threads 64..127 duplicate A chunks
    const int ar = ca >> 2, aseg = ca & 3;

    auto stage = [&](int buf, int k0) {
        async_cp16(&As[buf][ar][aseg * 8],
                   Aptr + (long long)(m0 + ar) * lda + k0 + aseg * 8);
#pragma unroll
        for (int it = 0; it < (BN * 4) / NT; ++it) {
            int c = tid + it * NT;
            int n = c >> 2, seg = c & 3;
            async_cp16(&Bsh[buf][n * 48 + seg * 8],
                       Bptr + (long long)(n0blk + n) * ldb + k0 + seg * 8);
        }
    };

    v8f acc = {0.f, 0.f, 0.f, 0.f, 0.f, 0.f, 0.f, 0.f};

    const int KT = K >> 5;
    stage(0, 0);
    for (int kt = 0; kt < KT; ++kt) {
        const int cur = kt & 1;
        if (kt + 1 < KT) {                   // overlap next tile's copies
            stage(cur ^ 1, (kt + 1) << 5);
            wait_async3();
        } else {
            wait_async0();
        }
        __syncthreads();                     // current tile fully in LDS

        // A fragment: element e -> K = (e/8)*16 + hs*8 + (e%8)
        Frag af;
        af.h[0] = *(const us8*)&As[cur][m][hs * 8];
        af.h[1] = *(const us8*)&As[cur][m][16 + hs * 8];
        // B fragment: element e -> K = (e/8)*16 + hs*8 + (e%8) at row n = nl
        Frag bf;
        bf.h[0] = *(const us8*)&Bsh[cur][nl * 48 + hs * 8];
        bf.h[1] = *(const us8*)&Bsh[cur][nl * 48 + 16 + hs * 8];

        acc = __builtin_amdgcn_wmma_f32_16x16x32_bf16(false, af.v, false, bf.v,
                                                      (short)0, acc, false, false);
        __syncthreads();                     // reads done before next overwrite
    }

    const int col = n0blk + nl;
    const float bv = bias ? bias[col] : 0.f;
#pragma unroll
    for (int j = 0; j < 8; j++) {
        int row = m0 + hs * 8 + j;
        float v = acc[j] * alpha + bv;
        if (VSTORE) {
            // V projection: row = b*LL + l, col = h*HD + jj -> Vt[((b*H+h)*HD+jj)*LL + l]
            int b = row / LL, l = row - b * LL;
            int h = col >> 5, jj = col & 31;
            Cbf[((long long)(b * HEADS + h) * HD + jj) * LL + l] = f2bf(v);
        } else {
            long long ci = (long long)blockIdx.z * sC + (long long)row * ldc + col;
            if (Cf)  Cf[ci]  = v;
            if (Cbf) Cbf[ci] = f2bf(v);
        }
    }
}

// =====================================================================
// transpose + convert: src [K,N] f32 -> dst [N,K] bf16
// =====================================================================
__global__ void transpose_f32_to_bf16_kernel(const float* __restrict__ src,
                                             __bf16* __restrict__ dst,
                                             int K, int N)
{
    int i = blockIdx.x * blockDim.x + threadIdx.x;
    if (i >= K * N) return;
    int k = i / N, n = i - k * N;
    dst[(long long)n * K + k] = f2bf(src[i]);
}

// =====================================================================
// text: emb gather -> TTFS encode -> combined[b, t*64+s, d] (bf16)
// =====================================================================
__global__ void text_encode_kernel(const int* __restrict__ tin,
                                   const float* __restrict__ emb,
                                   __bf16* __restrict__ comb)
{
    int i = blockIdx.x * blockDim.x + threadIdx.x;     // B*S_TEXT*D = 65536
    if (i >= BB * S_TEXT * DD) return;
    int d = i & (DD - 1);
    int s = (i >> 8) & (S_TEXT - 1);
    int b = i >> 14;
    int tok = tin[b * S_TEXT + s];
    float x = emb[(long long)tok * DD + d];
    float u = 1.f / (1.f + expf(-x));
    int ti = (int)floorf((1.f - u) * (float)(TT - 1));
    ti = ti < 0 ? 0 : (ti > TT - 1 ? TT - 1 : ti);
    for (int t = 0; t < TT; t++) {
        long long l = (long long)t * S_TEXT + s;
        comb[((long long)b * LL + l) * DD + d] = f2bf(t == ti ? x : 0.f);
    }
}

// =====================================================================
// audio: 2-layer MLP -> TTFS encode -> combined[b, 1280 + t*32+s, d]
// =====================================================================
__global__ void audio_encode_kernel(const float* __restrict__ audio,
                                    const float* __restrict__ aw1, const float* __restrict__ ab1,
                                    const float* __restrict__ aw2, const float* __restrict__ ab2,
                                    __bf16* __restrict__ comb)
{
    __shared__ float ain[AUDIO_F];
    __shared__ float hid[DD / 2];
    int b = blockIdx.x >> 5;
    int s = blockIdx.x & 31;
    int tid = threadIdx.x;
    if (tid < AUDIO_F) ain[tid] = audio[((long long)b * S_AUDIO + s) * AUDIO_F + tid];
    __syncthreads();
    if (tid < DD / 2) {
        float acc = ab1[tid];
        for (int f = 0; f < AUDIO_F; f++) acc += ain[f] * aw1[f * (DD / 2) + tid];
        hid[tid] = fmaxf(acc, 0.f);
    }
    __syncthreads();
    float p = ab2[tid];
    for (int j = 0; j < DD / 2; j++) p += hid[j] * aw2[j * DD + tid];
    float u = 1.f / (1.f + expf(-p));
    int ti = (int)floorf((1.f - u) * (float)(TT - 1));
    ti = ti < 0 ? 0 : (ti > TT - 1 ? TT - 1 : ti);
    for (int t = 0; t < TT; t++) {
        long long l = (long long)L_TEXT + t * S_AUDIO + s;
        comb[((long long)b * LL + l) * DD + tid] = f2bf(t == ti ? p : 0.f);
    }
}

// =====================================================================
// row softmax (f32 scores -> bf16 probs); one block of 256 per row
// =====================================================================
__global__ void softmax_bf16_kernel(const float* __restrict__ S,
                                    __bf16* __restrict__ P, int N)
{
    __shared__ float red[256];
    int row = blockIdx.x, tid = threadIdx.x;
    const float* x = S + (long long)row * N;
    __bf16* p = P + (long long)row * N;
    float mx = -3.4e38f;
    for (int c = tid; c < N; c += 256) mx = fmaxf(mx, x[c]);
    red[tid] = mx; __syncthreads();
    for (int s = 128; s > 0; s >>= 1) { if (tid < s) red[tid] = fmaxf(red[tid], red[tid + s]); __syncthreads(); }
    mx = red[0]; __syncthreads();
    float sum = 0.f;
    for (int c = tid; c < N; c += 256) sum += expf(x[c] - mx);
    red[tid] = sum; __syncthreads();
    for (int s = 128; s > 0; s >>= 1) { if (tid < s) red[tid] += red[tid + s]; __syncthreads(); }
    float inv = 1.f / red[0];
    for (int c = tid; c < N; c += 256) p[c] = f2bf(expf(x[c] - mx) * inv);
}

// =====================================================================
// temporal diff threshold scan (carry = previous *compressed* value)
// fused [B, l=t*64+s, D] (f32) -> comp [T,B,S,D] (f32)
// =====================================================================
__global__ void temporal_compress_kernel(const float* __restrict__ fused,
                                         float* __restrict__ comp)
{
    int i = blockIdx.x * blockDim.x + threadIdx.x;
    if (i >= BB * S_TEXT * DD) return;
    int d = i & (DD - 1);
    int s = (i >> 8) & (S_TEXT - 1);
    int b = i >> 14;
    float prev = fused[((long long)b * L_TEXT + s) * DD + d];
    comp[(((long long)0 * BB + b) * S_TEXT + s) * DD + d] = prev;
    for (int t = 1; t < TT; t++) {
        float cur = fused[((long long)b * L_TEXT + t * S_TEXT + s) * DD + d];
        float o = (fabsf(cur - prev) > 0.1f) ? cur : 0.f;
        comp[(((long long)t * BB + b) * S_TEXT + s) * DD + d] = o;
        prev = o;
    }
}

// =====================================================================
// per-timestep top-k sparsify via histogram threshold; one block per t
// =====================================================================
__global__ void __launch_bounds__(1024)
sparsify_kernel(const float* __restrict__ comp, __bf16* __restrict__ out,
                int perT, int kTop)
{
    __shared__ float rmax[1024];
    __shared__ float rsum[1024];
    __shared__ int   hist[1024];
    __shared__ float sh_max, sh_sum, sh_th;
    int t = blockIdx.x, tid = threadIdx.x;
    const float* x = comp + (long long)t * perT;
    __bf16* o = out + (long long)t * perT;

    float mx = 0.f, sm = 0.f;
    for (int i = tid; i < perT; i += 1024) { float v = x[i]; mx = fmaxf(mx, fabsf(v)); sm += v; }
    rmax[tid] = mx; rsum[tid] = sm; hist[tid] = 0;
    __syncthreads();
    for (int s = 512; s > 0; s >>= 1) {
        if (tid < s) { rmax[tid] = fmaxf(rmax[tid], rmax[tid + s]); rsum[tid] += rsum[tid + s]; }
        __syncthreads();
    }
    if (tid == 0) { sh_max = rmax[0]; sh_sum = rsum[0]; }
    __syncthreads();
    float maxabs = sh_max;
    if (maxabs > 0.f) {
        float invm = 1024.f / maxabs;
        for (int i = tid; i < perT; i += 1024) {
            float a = fabsf(x[i]);
            if (a > 0.f) { int bin = (int)(a * invm); bin = bin > 1023 ? 1023 : bin; atomicAdd(&hist[bin], 1); }
        }
    }
    __syncthreads();
    if (tid == 0) {
        float th = 0.f;
        if (maxabs > 0.f) {
            int acc = 0;
            for (int b = 1023; b >= 0; b--) {
                acc += hist[b];
                if (acc >= kTop) { th = (float)b * maxabs * (1.f / 1024.f); break; }
            }
        }
        sh_th = th;
    }
    __syncthreads();
    float th = sh_th;
    bool keep = sh_sum > 0.f;
    for (int i = tid; i < perT; i += 1024) {
        float v = x[i];
        float r = keep ? ((fabsf(v) >= th) ? v : 0.f) : v;
        o[i] = f2bf(r);
    }
}

// =====================================================================
// diagonal SSM scan over T; u,h layout [T,B,S,DSTATE]
// =====================================================================
__global__ void ssm_scan_kernel(const float* __restrict__ u,
                                const float* __restrict__ Arow,
                                __bf16* __restrict__ h)
{
    int i = blockIdx.x * blockDim.x + threadIdx.x;
    if (i >= BB * S_TEXT * DSTATE) return;
    int n = i & (DSTATE - 1);
    int s = (i >> 6) & (S_TEXT - 1);
    int b = i >> 12;
    float a = Arow[n];
    float st = 0.f;
    for (int t = 0; t < TT; t++) {
        long long idx = (((long long)t * BB + b) * S_TEXT + s) * DSTATE + n;
        st = st * a + u[idx];
        h[idx] = f2bf(st);
    }
}

// =====================================================================
// mean over T: y [T,B,S,D] bf16 -> feat [B*S, D] bf16
// =====================================================================
__global__ void mean_time_kernel(const __bf16* __restrict__ y,
                                 __bf16* __restrict__ feat)
{
    int i = blockIdx.x * blockDim.x + threadIdx.x;
    if (i >= BB * S_TEXT * DD) return;
    int d = i & (DD - 1);
    int s = (i >> 8) & (S_TEXT - 1);
    int b = i >> 14;
    float acc = 0.f;
    for (int t = 0; t < TT; t++)
        acc += bf2f(y[(((long long)t * BB + b) * S_TEXT + s) * DD + d]);
    feat[((long long)b * S_TEXT + s) * DD + d] = f2bf(acc * (1.f / (float)TT));
}

// =====================================================================
// host-side orchestration
// =====================================================================
extern "C" void kernel_launch(void* const* d_in, const int* in_sizes, int n_in,
                              void* d_out, int out_size, void* d_ws, size_t ws_size,
                              hipStream_t stream)
{
    (void)in_sizes; (void)n_in; (void)out_size; (void)ws_size;
    const int*   tin   = (const int*)  d_in[0];
    const float* audio = (const float*)d_in[1];
    const float* emb   = (const float*)d_in[2];
    const float* aw1   = (const float*)d_in[3];
    const float* ab1   = (const float*)d_in[4];
    const float* aw2   = (const float*)d_in[5];
    const float* ab2   = (const float*)d_in[6];
    const float* wq    = (const float*)d_in[7];
    const float* bq    = (const float*)d_in[8];
    const float* wk    = (const float*)d_in[9];
    const float* bk    = (const float*)d_in[10];
    const float* wv    = (const float*)d_in[11];
    const float* bv    = (const float*)d_in[12];
    const float* wo    = (const float*)d_in[13];
    const float* bo    = (const float*)d_in[14];
    const float* Afp   = (const float*)d_in[15];
    const float* Bm    = (const float*)d_in[16];
    const float* Cm    = (const float*)d_in[17];
    const float* wout  = (const float*)d_in[18];
    const float* bout  = (const float*)d_in[19];
    float* outp = (float*)d_out;

    size_t off = 0;
    auto alloc = [&](size_t bytes) -> void* {
        void* p = (char*)d_ws + off;
        off += (bytes + 255) & ~(size_t)255;
        return p;
    };
    __bf16* combBF  = (__bf16*)alloc((size_t)BB * LL * DD * 2);
    __bf16* wqT     = (__bf16*)alloc((size_t)DD * DD * 2);         // [N,K]
    __bf16* wkT     = (__bf16*)alloc((size_t)DD * DD * 2);
    __bf16* wvT     = (__bf16*)alloc((size_t)DD * DD * 2);
    __bf16* woT     = (__bf16*)alloc((size_t)DD * DD * 2);
    __bf16* BmT     = (__bf16*)alloc((size_t)3 * DSTATE * DD * 2); // per layer [64,256]
    __bf16* CmT     = (__bf16*)alloc((size_t)3 * DD * DSTATE * 2); // per layer [256,64]
    __bf16* woutT   = (__bf16*)alloc((size_t)VOCAB * DD * 2);      // [32000,256]
    __bf16* QBF     = (__bf16*)alloc((size_t)BB * LL * DD * 2);
    __bf16* KBF     = (__bf16*)alloc((size_t)BB * LL * DD * 2);
    __bf16* VtBF    = (__bf16*)alloc((size_t)BB * HEADS * HD * LL * 2);  // head-transposed V
    float*  scoresF = (float*) alloc((size_t)LL * LL * 4);
    __bf16* probsBF = (__bf16*)alloc((size_t)LL * LL * 2);
    __bf16* attBF   = (__bf16*)alloc((size_t)BB * LL * DD * 2);
    float*  fusedF  = (float*) alloc((size_t)BB * L_TEXT * DD * 4);
    float*  compF   = (float*) alloc((size_t)TT * BB * S_TEXT * DD * 4);
    __bf16* xBF     = (__bf16*)alloc((size_t)TT * BB * S_TEXT * DD * 2);
    float*  uF      = (float*) alloc((size_t)TT * BB * S_TEXT * DSTATE * 4);
    __bf16* hstBF   = (__bf16*)alloc((size_t)TT * BB * S_TEXT * DSTATE * 2);
    __bf16* featBF  = (__bf16*)alloc((size_t)BB * S_TEXT * DD * 2);

    auto transConv = [&](const float* s, __bf16* d, int K, int N) {
        transpose_f32_to_bf16_kernel<<<(K * N + 255) / 256, 256, 0, stream>>>(s, d, K, N);
    };
    transConv(wq, wqT, DD, DD);
    transConv(wk, wkT, DD, DD);
    transConv(wv, wvT, DD, DD);
    transConv(wo, woT, DD, DD);
    for (int i = 0; i < 3; i++) {
        transConv(Bm + (long long)i * DD * DSTATE, BmT + (long long)i * DSTATE * DD, DD, DSTATE);
        transConv(Cm + (long long)i * DSTATE * DD, CmT + (long long)i * DD * DSTATE, DSTATE, DD);
    }
    transConv(wout, woutT, DD, VOCAB);

    // 1) spike encoding -> combined bf16 [B, L, D]
    text_encode_kernel<<<(BB * S_TEXT * DD + 255) / 256, 256, 0, stream>>>(tin, emb, combBF);
    audio_encode_kernel<<<BB * S_AUDIO, 256, 0, stream>>>(audio, aw1, ab1, aw2, ab2, combBF);

    // 2) QKV projections: [7680,256] @ Wt[256,256]; V stored head-transposed
    {
        dim3 g(BB * LL / 16, DD / 64, 1);
        gemm_bf16_wmma<64, 0><<<g, 128, 0, stream>>>(combBF, wqT, nullptr, QBF, bq,
            DD, DD, DD, DD, 0, 0, 0, 1.f);
        gemm_bf16_wmma<64, 0><<<g, 128, 0, stream>>>(combBF, wkT, nullptr, KBF, bk,
            DD, DD, DD, DD, 0, 0, 0, 1.f);
        gemm_bf16_wmma<64, 1><<<g, 128, 0, stream>>>(combBF, wvT, nullptr, VtBF, bv,
            DD, DD, DD, DD, 0, 0, 0, 1.f);
    }

    // 3) attention per (b,h): S = (Q Kt)/sqrt(hd); P = softmax(S); att = P V
    const float scale = 0.17677669529663687f;   // 1/sqrt(32)
    for (int bh = 0; bh < BB * HEADS; bh++) {
        int b = bh / HEADS, h = bh % HEADS;
        const __bf16* Qh  = QBF  + (long long)b * LL * DD + h * HD;
        const __bf16* Kh  = KBF  + (long long)b * LL * DD + h * HD;
        const __bf16* Vth = VtBF + (long long)(b * HEADS + h) * HD * LL;   // [32, 1920]
        __bf16* atth = attBF + (long long)b * LL * DD + h * HD;
        // scores: M=L, N=L(keys), K=hd; B rows are keys with ldb=DD
        gemm_bf16_wmma<64, 0><<<dim3(LL / 16, LL / 64, 1), 128, 0, stream>>>(
            Qh, Kh, scoresF, nullptr, nullptr,
            HD, DD, DD, LL, 0, 0, 0, scale);
        softmax_bf16_kernel<<<LL, 256, 0, stream>>>(scoresF, probsBF, LL);
        // att = P @ V: M=L, N=hd, K=L; B = Vt [32,1920]
        gemm_bf16_wmma<32, 0><<<dim3(LL / 16, 1, 1), 64, 0, stream>>>(
            probsBF, Vth, nullptr, atth, nullptr,
            LL, LL, LL, DD, 0, 0, 0, 1.f);
    }

    // 4) fused = att[:, :1280] @ wo + bo  (batched over b)
    gemm_bf16_wmma<64, 0><<<dim3(L_TEXT / 16, DD / 64, BB), 128, 0, stream>>>(
        attBF, woT, fusedF, nullptr, bo,
        DD, DD, DD, DD,
        (long long)LL * DD, 0, (long long)L_TEXT * DD, 1.f);

    // 5) spike-pattern optimization
    temporal_compress_kernel<<<(BB * S_TEXT * DD + 255) / 256, 256, 0, stream>>>(fusedF, compF);
    sparsify_kernel<<<TT, 1024, 0, stream>>>(compF, xBF, BB * S_TEXT * DD,
                                             (int)(BB * S_TEXT * DD * 0.3));

    // 6) SSM stack: u = x@Bp; scan; y = h@Cp
    for (int i = 0; i < 3; i++) {
        gemm_bf16_wmma<64, 0><<<dim3(TT * BB * S_TEXT / 16, 1, 1), 128, 0, stream>>>(
            xBF, BmT + (long long)i * DSTATE * DD, uF, nullptr, nullptr,
            DD, DD, DD, DSTATE, 0, 0, 0, 1.f);
        ssm_scan_kernel<<<(BB * S_TEXT * DSTATE + 255) / 256, 256, 0, stream>>>(
            uF, Afp + i * DSTATE, hstBF);
        gemm_bf16_wmma<64, 0><<<dim3(TT * BB * S_TEXT / 16, DD / 64, 1), 128, 0, stream>>>(
            hstBF, CmT + (long long)i * DD * DSTATE, nullptr, xBF, nullptr,
            DSTATE, DSTATE, DSTATE, DD, 0, 0, 0, 1.f);
    }

    // 7) mean over T -> head GEMM into d_out
    mean_time_kernel<<<(BB * S_TEXT * DD + 255) / 256, 256, 0, stream>>>(xBF, featBF);
    gemm_bf16_wmma<64, 0><<<dim3(BB * S_TEXT / 16, VOCAB / 64, 1), 128, 0, stream>>>(
        featBF, woutT, outp, nullptr, bout,
        DD, DD, DD, VOCAB, 0, 0, 0, 1.f);
}
